// BoxCountingDimensionLoss_3977139716657
// MI455X (gfx1250) — matrix operand order repaired and checked by reading, not compiled, not run on hardware
//
#include <hip/hip_runtime.h>
#include <hip/hip_bf16.h>

// ---------------------------------------------------------------------------
// BoxCountingDimensionLoss for MI455X (gfx1250, wave32, WMMA)
//   B=8, N=2048, D=64.  Fused symmetric Gram kernel: each wave computes a
//   32x32 output block (2x2 WMMA tiles, 8x v_wmma_f32_16x16x32_f16, K=64),
//   then the 8-epsilon exp2 + sqrt epilogue on registers using raw
//   v_exp_f32 / v_sqrt_f32 (no libm range fixup).  Upper triangle only,
//   weight 2 off-diagonal blocks.
// ---------------------------------------------------------------------------

#define NB    8
#define NPTS  2048
#define DIM   64
#define NT2   64                    // 32-wide blocks per dimension
#define PPB2  (NT2 * (NT2 + 1) / 2) // 2080 upper-triangular block pairs/batch

typedef __attribute__((ext_vector_type(16))) _Float16 v16h;
typedef __attribute__((ext_vector_type(8)))  float    v8f;

__device__ __forceinline__ int tri_off2(int it) {
  return (it * (2 * NT2 + 1 - it)) >> 1;
}

// ---------------- init: zero accumulators + epsilon coefficients -----------
__global__ void bcd_init(const float* __restrict__ eps, float* __restrict__ accum) {
  const int t = threadIdx.x;
  if (t < 11) accum[t] = 0.0f;
  if (t >= 16 && t < 24) {
    float ee = eps[t - 16];
    // exp(-sq/(2*sigma^2*eps^2)) = exp2(-sq * 50*log2(e) / eps^2)
    accum[t] = (50.0f * 1.4426950408889634f) / (ee * ee);
  }
}

// ----------------- per-row stats: sq-norms, ltz, add-to-one ----------------
__global__ void bcd_row_stats(const float* __restrict__ pts,
                              float* __restrict__ sqn,
                              float* __restrict__ accum) {
  const int row = blockIdx.x * blockDim.x + threadIdx.x;   // 64 * 256 = 16384
  const float* p = pts + (size_t)row * DIM;
  float sn = 0.f, rs = 0.f, ltz = 0.f;
#pragma unroll
  for (int i = 0; i < DIM; ++i) {
    float x = p[i];
    sn = fmaf(x, x, sn);
    rs += x;
    float nx = fminf(x, 0.0f);          // relu(-x) = -min(x,0)
    ltz = fmaf(nx, nx, ltz);
  }
  sqn[row] = sn;
  float ato = (rs - 1.0f) * (rs - 1.0f);
#pragma unroll
  for (int o = 16; o > 0; o >>= 1) {
    ltz += __shfl_xor(ltz, o);
    ato += __shfl_xor(ato, o);
  }
  __shared__ float red[16];
  const int lane = threadIdx.x & 31, wid = threadIdx.x >> 5;
  if (lane == 0) { red[wid * 2] = ltz; red[wid * 2 + 1] = ato; }
  __syncthreads();
  if (threadIdx.x < 2) {
    float s = 0.f;
    for (int w2 = 0; w2 < 8; ++w2) s += red[w2 * 2 + threadIdx.x];
    atomicAdd(&accum[9 + threadIdx.x], s);
  }
}

// ----------------------- operand gathers (fp32 -> f16) ---------------------
// B operand 32x16 f16, contiguous k order (unused): lane l -> column n=l&15, K = 16*(l>>4) + [0..16)
__device__ __forceinline__ v16h cvtB(const float* __restrict__ p) {
  v16h r;
#pragma unroll
  for (int i = 0; i < 16; ++i) r[i] = (_Float16)p[i];
  return r;
}
// A operand 16x32 f16: lane l -> row m=l&15, halves 0..7 = K k0+8*hs+[0..8),
// halves 8..15 = K k0+16+8*hs+[0..8)  (hs = l>>4).
__device__ __forceinline__ v16h cvtA(const float* __restrict__ row, int k0, int hs) {
  v16h r;
  const float* p0 = row + k0 + 8 * hs;
  const float* p1 = row + k0 + 16 + 8 * hs;
#pragma unroll
  for (int i = 0; i < 8; ++i) {
    r[i]     = (_Float16)p0[i];
    r[8 + i] = (_Float16)p1[i];
  }
  return r;
}

// epilogue for one 16x16 tile held in a v8f accumulator
__device__ __forceinline__ void epi_tile(const v8f& c, const float* __restrict__ sni,
                                         float snj, const float* __restrict__ ce2,
                                         float* __restrict__ vals) {
#pragma unroll
  for (int r = 0; r < 8; ++r) {
    float sq = fmaxf(sni[r] + snj - 2.0f * c[r], 0.0f);
    vals[8] += __builtin_amdgcn_sqrtf(sq);            // raw v_sqrt_f32
#pragma unroll
    for (int e = 0; e < 8; ++e)
      vals[e] += __builtin_amdgcn_exp2f(-sq * ce2[e]); // raw v_exp_f32
  }
}

// ------------------------- main fused Gram-block kernel --------------------
__global__ void __launch_bounds__(256, 2)
bcd_gram(const float* __restrict__ pts,
         const float* __restrict__ sqn,
         float* __restrict__ accum) {
  const int lane = threadIdx.x & 31;
  const int wid  = threadIdx.x >> 5;
  const int w    = blockIdx.x * 8 + wid;        // 2080 blocks * 8 waves = 16640
  const int b    = w / PPB2;
  const int t    = w - b * PPB2;

  // decode triangular block index t -> (it <= jt), 64x64 block grid
  int it = (int)(64.5f - __builtin_amdgcn_sqrtf(fmaxf(4160.25f - 2.0f * (float)t, 0.0f)));
  it = it < 0 ? 0 : (it > NT2 - 1 ? NT2 - 1 : it);
  while (it < NT2 - 1 && tri_off2(it + 1) <= t) ++it;
  while (it > 0 && tri_off2(it) > t) --it;
  const int jt = it + (t - tri_off2(it));
  const int i0 = it << 5, j0 = jt << 5;         // 32-wide blocks
  const float wgt = (it == jt) ? 1.0f : 2.0f;   // diagonal block holds its own mirror

  // uniform per-epsilon coefficients (precomputed in bcd_init)
  float ce2[8];
#pragma unroll
  for (int e = 0; e < 8; ++e) ce2[e] = accum[16 + e];

  const int m = lane & 15, hs = lane >> 4;
  const float* base  = pts + (size_t)b * (NPTS * DIM);
  const float* arow0 = base + (size_t)(i0 + m) * DIM;
  const float* arow1 = base + (size_t)(i0 + 16 + m) * DIM;
  const float* brow0 = base + (size_t)(j0 + m) * DIM;
  const float* brow1 = base + (size_t)(j0 + 16 + m) * DIM;

  v8f c00 = {}, c01 = {}, c10 = {}, c11 = {};
#pragma unroll
  for (int k0 = 0; k0 < DIM; k0 += 32) {
    v16h a0 = cvtA(arow0, k0, hs);
    v16h a1 = cvtA(arow1, k0, hs);
    v16h b0 = cvtA(brow0, k0, hs);
    v16h b1 = cvtA(brow1, k0, hs);
    // 4 independent WMMAs: operand reuse x2, no D->A/B hazards in between
    c00 = __builtin_amdgcn_wmma_f32_16x16x32_f16(false, a0, false, b0, (short)0, c00, false, false);
    c01 = __builtin_amdgcn_wmma_f32_16x16x32_f16(false, a0, false, b1, (short)0, c01, false, false);
    c10 = __builtin_amdgcn_wmma_f32_16x16x32_f16(false, a1, false, b0, (short)0, c10, false, false);
    c11 = __builtin_amdgcn_wmma_f32_16x16x32_f16(false, a1, false, b1, (short)0, c11, false, false);
  }

  // epilogue: C layout -> VGPR r holds (M = r + 8*hs, N = lane&15)
  const float* sqb = sqn + b * NPTS;
  const float snj0 = sqb[j0 + m];
  const float snj1 = sqb[j0 + 16 + m];
  float sni0[8], sni1[8];
#pragma unroll
  for (int r = 0; r < 8; ++r) {
    sni0[r] = sqb[i0 + 8 * hs + r];
    sni1[r] = sqb[i0 + 16 + 8 * hs + r];
  }

  float vals[9];
#pragma unroll
  for (int k = 0; k < 9; ++k) vals[k] = 0.f;
  epi_tile(c00, sni0, snj0, ce2, vals);
  epi_tile(c01, sni0, snj1, ce2, vals);
  epi_tile(c10, sni1, snj0, ce2, vals);
  epi_tile(c11, sni1, snj1, ce2, vals);

  // weight + wave32 shuffle reduction
#pragma unroll
  for (int k = 0; k < 9; ++k) {
    vals[k] *= wgt;
#pragma unroll
    for (int o = 16; o > 0; o >>= 1) vals[k] += __shfl_xor(vals[k], o);
  }

  __shared__ float red[8 * 9];
  if (lane == 0) {
#pragma unroll
    for (int k = 0; k < 9; ++k) red[wid * 9 + k] = vals[k];
  }
  __syncthreads();
  if (threadIdx.x < 9) {
    float s = 0.f;
#pragma unroll
    for (int w2 = 0; w2 < 8; ++w2) s += red[w2 * 9 + threadIdx.x];
    atomicAdd(&accum[threadIdx.x], s);
  }
}

// -------------------- finalize: log-log fit + combine ----------------------
__global__ void bcd_finalize(const float* __restrict__ eps,
                             const float* __restrict__ accum,
                             float* __restrict__ out) {
  float Sx = 0.f, Sy = 0.f, Sxx = 0.f, Sxy = 0.f;
  for (int e = 0; e < 8; ++e) {
    float le = logf(eps[e]);
    float lc = logf(accum[e] * (1.0f / 16384.0f));   // counts_e
    Sx += le; Sy += lc; Sxx += le * le; Sxy += le * lc;
  }
  float slope  = (8.0f * Sxy - Sx * Sy) / (8.0f * Sxx - Sx * Sx);
  float spread = accum[8]  * (1.0f / (8.0f * 2048.0f * 2048.0f));
  float ltz    = accum[9]  * (1.0f / (8.0f * 2048.0f * 64.0f));
  float ato    = accum[10] * (1.0f / 16384.0f);
  out[0] = slope - 0.1f * spread + 0.1f * ltz + 0.1f * ato;
}

// ---------------------------------------------------------------------------
extern "C" void kernel_launch(void* const* d_in, const int* in_sizes, int n_in,
                              void* d_out, int out_size, void* d_ws, size_t ws_size,
                              hipStream_t stream) {
  (void)in_sizes; (void)n_in; (void)out_size; (void)ws_size;
  const float* pts = (const float*)d_in[0];   // [8,2048,64] fp32
  const float* eps = (const float*)d_in[1];   // [8] fp32
  float* out   = (float*)d_out;               // scalar fp32
  float* accum = (float*)d_ws;                        // 24 floats (pad to 256B)
  float* sqn   = (float*)((char*)d_ws + 256);         // 16384 floats (64 KB)

  bcd_init<<<1, 32, 0, stream>>>(eps, accum);
  bcd_row_stats<<<(NB * NPTS) / 256, 256, 0, stream>>>(pts, sqn, accum);
  bcd_gram<<<(NB * PPB2) / 8, 256, 0, stream>>>(pts, sqn, accum);
  bcd_finalize<<<1, 1, 0, stream>>>(eps, accum, out);
}
